// DGCNNSeg_74826920231286
// MI455X (gfx1250) — matrix operand order repaired and checked by reading, not compiled, not run on hardware
//
#include <hip/hip_runtime.h>
#include <hip/hip_bf16.h>
#include <math.h>

typedef __attribute__((ext_vector_type(16))) __bf16 v16bf;
typedef __attribute__((ext_vector_type(8)))  __bf16 v8bf;
typedef __attribute__((ext_vector_type(8)))  float  v8f;

#define KNN_K 20

// ---------------------------------------------------------------------------
// Fragment loaders (CDNA5 ISA 7.12.2 layouts, packed bf16 rows, Kp % 32 == 0)
//   A lane(m=lane&15): K runs [8h,8h+8) and [16+8h,16+8h+8)   (h = lane>>4)
//   B lane(n=lane&15): K runs [8h,8h+8) and [16+8h,16+8h+8)   (h = lane>>4)
// ---------------------------------------------------------------------------
__device__ __forceinline__ v16bf load_a_frag(const __bf16* row, int kk, int aoff0, int aoff1)
{
    v8bf x0 = *(const v8bf*)(row + kk + aoff0);
    v8bf x1 = *(const v8bf*)(row + kk + aoff1);
    return __builtin_shufflevector(x0, x1, 0,1,2,3,4,5,6,7,8,9,10,11,12,13,14,15);
}
__device__ __forceinline__ v16bf load_b_frag(const __bf16* row, int kk, int boff)
{
    v8bf x0 = *(const v8bf*)(row + kk + boff);
    v8bf x1 = *(const v8bf*)(row + kk + boff + 16);
    return __builtin_shufflevector(x0, x1, 0,1,2,3,4,5,6,7,8,9,10,11,12,13,14,15);
}

__device__ __forceinline__ void store_tile16(const v8f& acc, float* __restrict__ C, int ldc,
                                             int cmbase, int cn, float scale, float shift, int relu)
{
    #pragma unroll
    for (int r = 0; r < 8; ++r) {
        float v = acc[r] * scale + shift;
        if (relu) v = fmaxf(v, 0.0f);
        C[(size_t)(cmbase + r) * ldc + cn] = v;
    }
}

// ---------------------------------------------------------------------------
// bf16 WMMA GEMM, 2x2 register-blocked: one wave32 computes a 32x32 tile.
//   A : [M , Kp] bf16 packed (M % 32 == 0, Kp % 32 == 0)
//   W : [Np, Kp] bf16 packed (Np % 32 == 0, zero rows beyond Nreal)
//   C[m][n] = sum_k A[m][k]*W[n][k]  (+ fused BN / bias / ReLU)
// Per K-step: 8x global_load_b128 -> 4x WMMA (each fragment reused twice).
// ---------------------------------------------------------------------------
__global__ __launch_bounds__(256)
void gemm_bf16_wmma(const __bf16* __restrict__ A,
                    const __bf16* __restrict__ W,
                    float* __restrict__ C, int ldc,
                    int M, int Nreal, int Np, int Kp,
                    const float* __restrict__ gam, const float* __restrict__ bet,
                    const float* __restrict__ mu,  const float* __restrict__ var,
                    const float* __restrict__ bias, int relu)
{
    const int lane = threadIdx.x & 31;
    const int wave = threadIdx.x >> 5;
    const int tilesN = Np >> 5;
    const int tilesM = M >> 5;
    const int tile = blockIdx.x * 8 + wave;       // wave-uniform -> EXEC all-1s at WMMA
    if (tile >= tilesM * tilesN) return;
    const int tm = tile / tilesN;                 // adjacent waves share tm -> share A lines
    const int tn = tile % tilesN;

    const int half = lane >> 4;
    const int l15  = lane & 15;

    const __bf16* Arow0 = A + (size_t)(tm * 32 + l15) * Kp;
    const __bf16* Arow1 = Arow0 + (size_t)16 * Kp;
    const __bf16* Wrow0 = W + (size_t)(tn * 32 + l15) * Kp;
    const __bf16* Wrow1 = Wrow0 + (size_t)16 * Kp;
    const int aoff0 = 8 * half;
    const int aoff1 = 16 + 8 * half;
    const int boff  = 8 * half;

    v8f acc00 = {}, acc01 = {}, acc10 = {}, acc11 = {};
    #pragma unroll 2
    for (int kk = 0; kk < Kp; kk += 32) {
        __builtin_prefetch((const void*)(Arow0 + kk + 64), 0, 3);
        __builtin_prefetch((const void*)(Arow1 + kk + 64), 0, 3);
        __builtin_prefetch((const void*)(Wrow0 + kk + 64), 0, 3);
        __builtin_prefetch((const void*)(Wrow1 + kk + 64), 0, 3);
        v16bf a0 = load_a_frag(Arow0, kk, aoff0, aoff1);
        v16bf a1 = load_a_frag(Arow1, kk, aoff0, aoff1);
        v16bf b0 = load_b_frag(Wrow0, kk, boff);
        v16bf b1 = load_b_frag(Wrow1, kk, boff);
        acc00 = __builtin_amdgcn_wmma_f32_16x16x32_bf16(false, a0, false, b0, (short)0, acc00, false, false);
        acc01 = __builtin_amdgcn_wmma_f32_16x16x32_bf16(false, a0, false, b1, (short)0, acc01, false, false);
        acc10 = __builtin_amdgcn_wmma_f32_16x16x32_bf16(false, a1, false, b0, (short)0, acc10, false, false);
        acc11 = __builtin_amdgcn_wmma_f32_16x16x32_bf16(false, a1, false, b1, (short)0, acc11, false, false);
    }

    // D lane layout: m = 16*i + 8*half + r, n = 16*j + l15 within the 32x32 tile
    const int cm0 = tm * 32 + 8 * half;
    #pragma unroll
    for (int j = 0; j < 2; ++j) {
        const int cn = tn * 32 + 16 * j + l15;
        if (cn < Nreal) {
            float scale = 1.0f, shift = 0.0f;
            if (gam) {
                float inv = gam[cn] * rsqrtf(var[cn] + 1e-5f);
                scale = inv;
                shift = bet[cn] - mu[cn] * inv;
            }
            if (bias) shift += bias[cn];
            if (j == 0) {
                store_tile16(acc00, C, ldc, cm0,      cn, scale, shift, relu);
                store_tile16(acc10, C, ldc, cm0 + 16, cn, scale, shift, relu);
            } else {
                store_tile16(acc01, C, ldc, cm0,      cn, scale, shift, relu);
                store_tile16(acc11, C, ldc, cm0 + 16, cn, scale, shift, relu);
            }
        }
    }
}

// ---------------------------------------------------------------------------
// Packing kernels: float (strided) -> bf16, zero-padded to (rows, Kp).
// ---------------------------------------------------------------------------
__global__ __launch_bounds__(256)
void pack_a_kernel(const float* __restrict__ A, int lda,
                   __bf16* __restrict__ Ap, int K, int Kp, int total /*M*Kp*/)
{
    int t = blockIdx.x * 256 + threadIdx.x;
    if (t >= total) return;
    int k = t % Kp;
    int m = t / Kp;
    float v = (k < K) ? A[(size_t)m * lda + k] : 0.0f;
    Ap[t] = (__bf16)v;
}

__global__ __launch_bounds__(256)
void pack_w_kernel(const float* __restrict__ W, int ldw, int colOff,
                   int nSrcRows, int rowOff,
                   __bf16* __restrict__ Wp, int K, int Kp, int total /*nOutRows*Kp*/)
{
    int t = blockIdx.x * 256 + threadIdx.x;
    if (t >= total) return;
    int k = t % Kp;
    int r = t / Kp;
    float v = (r < nSrcRows && k < K) ? W[(size_t)r * ldw + colOff + k] : 0.0f;
    Wp[(size_t)(rowOff + r) * Kp + k] = (__bf16)v;
}

// ---------------------------------------------------------------------------
// kNN: per query, top-20 smallest squared distances (self excluded).
// 64 queries/block; query chans register-blocked (4 at a time); 32 candidate
// accumulators in registers -> ~1 ds_load per 2 VALU (VALU-bound).
// Channels zero-padded to a multiple of 4 in LDS.
// ---------------------------------------------------------------------------
__global__ __launch_bounds__(64)
void knn_kernel(const float* __restrict__ x, int xstride, int C,
                int Nn, int* __restrict__ idx)
{
    __shared__ float shQ[64][129];
    __shared__ float shC[32][129];
    const int tid = threadIdx.x;
    const int blocksPerBatch = Nn / 64;
    const int b = blockIdx.x / blocksPerBatch;
    const int qbase = (blockIdx.x % blocksPerBatch) * 64;
    const float* xb = x + (size_t)b * Nn * xstride;
    const int Cp = (C + 3) & ~3;

    for (int i = tid; i < 64 * Cp; i += 64) {
        int r = i / Cp, c = i - r * Cp;
        shQ[r][c] = (c < C) ? xb[(size_t)(qbase + r) * xstride + c] : 0.0f;
    }
    __syncthreads();

    const int q = qbase + tid;
    float bd[KNN_K];
    int   bi[KNN_K];
    #pragma unroll
    for (int j = 0; j < KNN_K; ++j) { bd[j] = 3.0e38f; bi[j] = 0; }

    for (int t = 0; t < Nn; t += 32) {
        for (int i = tid; i < 32 * Cp; i += 64) {
            int r = i / Cp, c = i - r * Cp;
            shC[r][c] = (c < C) ? xb[(size_t)(t + r) * xstride + c] : 0.0f;
        }
        __syncthreads();

        float dacc[32];
        #pragma unroll
        for (int j = 0; j < 32; ++j) dacc[j] = 0.0f;

        for (int c0 = 0; c0 < Cp; c0 += 4) {
            float q0 = shQ[tid][c0 + 0];
            float q1 = shQ[tid][c0 + 1];
            float q2 = shQ[tid][c0 + 2];
            float q3 = shQ[tid][c0 + 3];
            #pragma unroll
            for (int j = 0; j < 32; ++j) {
                float d0 = q0 - shC[j][c0 + 0];
                float d1 = q1 - shC[j][c0 + 1];
                float d2 = q2 - shC[j][c0 + 2];
                float d3 = q3 - shC[j][c0 + 3];
                dacc[j] = fmaf(d0, d0, fmaf(d1, d1, fmaf(d2, d2, fmaf(d3, d3, dacc[j]))));
            }
        }

        #pragma unroll
        for (int j = 0; j < 32; ++j) {
            int m = t + j;
            float d = dacc[j];
            if (m != q && d < bd[KNN_K - 1]) {
                float cd = d; int ci = m;
                #pragma unroll
                for (int jj = 0; jj < KNN_K; ++jj) {
                    if (cd < bd[jj]) {
                        float td = bd[jj]; bd[jj] = cd; cd = td;
                        int   ti = bi[jj]; bi[jj] = ci; ci = ti;
                    }
                }
            }
        }
        __syncthreads();
    }
    int* orow = idx + ((size_t)b * Nn + q) * KNN_K;
    #pragma unroll
    for (int j = 0; j < KNN_K; ++j) orow[j] = bi[j];
}

// ---------------------------------------------------------------------------
// Edge epilogue: out[n,o] = relu(bn( max_k( y[idx[n,k]] - y[n] + z[n] ) ))
// yz rows = [y(Co) | z(Co)], row stride 2*Co.
// ---------------------------------------------------------------------------
__global__ __launch_bounds__(256)
void edge_combine(const float* __restrict__ yz, const int* __restrict__ idx,
                  const float* __restrict__ gam, const float* __restrict__ bet,
                  const float* __restrict__ mu,  const float* __restrict__ var,
                  float* __restrict__ out, int ldo, int Nn, int Co, int total)
{
    int t = blockIdx.x * 256 + threadIdx.x;
    if (t >= total) return;
    int o  = t % Co;
    int rn = t / Co;                 // b*Nn + n
    int b  = rn / Nn;
    int nl = rn - b * Nn;
    const int ldyz = 2 * Co;
    const float* yzb = yz + (size_t)b * Nn * ldyz;

    float yc = yzb[(size_t)nl * ldyz + o];
    float zc = yzb[(size_t)nl * ldyz + Co + o];
    float base = zc - yc;

    const int* irow = idx + (size_t)rn * KNN_K;
    float m = -3.0e38f;
    #pragma unroll
    for (int kk = 0; kk < KNN_K; ++kk) {
        int g = irow[kk];
        float yv = yzb[(size_t)g * ldyz + o];
        m = fmaxf(m, yv + base);
    }
    float inv = gam[o] * rsqrtf(var[o] + 1e-5f);
    float v = (m - mu[o]) * inv + bet[o];
    out[(size_t)rn * ldo + o] = fmaxf(v, 0.0f);
}

// Column-wise max over the N points of each batch (input already BN+ReLU'd).
__global__ __launch_bounds__(256)
void colmax_kernel(const float* __restrict__ g, float* __restrict__ gmax,
                   int Nn, int C, int total)
{
    int t = blockIdx.x * 256 + threadIdx.x;
    if (t >= total) return;
    int b = t / C, c = t - b * C;
    const float* gb = g + (size_t)b * Nn * C + c;
    float m = -3.0e38f;
    for (int n = 0; n < Nn; ++n) m = fmaxf(m, gb[(size_t)n * C]);
    gmax[t] = m;
}

// feat[rn] = [ xcat[rn] (512) | gmax[b] (1024) ]
__global__ __launch_bounds__(256)
void build_feat_kernel(const float* __restrict__ xcat, const float* __restrict__ gmax,
                       float* __restrict__ feat, int Nn, int total)
{
    int t = blockIdx.x * 256 + threadIdx.x;
    if (t >= total) return;
    int j  = t % 1536;
    int rn = t / 1536;
    int b  = rn / Nn;
    float v = (j < 512) ? xcat[(size_t)rn * 512 + j] : gmax[b * 1024 + (j - 512)];
    feat[(size_t)rn * 1536 + j] = v;
}

// ---------------------------------------------------------------------------
static inline void launch_gemm(const __bf16* Ap, const __bf16* Wp,
                               float* C, int ldc, int M, int Nreal, int Np, int Kp,
                               const float* g, const float* b, const float* m, const float* v,
                               const float* bias, int relu, hipStream_t s)
{
    int tiles = (M >> 5) * (Np >> 5);
    int blocks = (tiles + 7) / 8;
    gemm_bf16_wmma<<<blocks, 256, 0, s>>>(Ap, Wp, C, ldc, M, Nreal, Np, Kp,
                                          g, b, m, v, bias, relu);
}

static inline void launch_pack_a(const float* A, int lda, __bf16* Ap,
                                 int K, int Kp, int M, hipStream_t s)
{
    int total = M * Kp;
    pack_a_kernel<<<(total + 255) / 256, 256, 0, s>>>(A, lda, Ap, K, Kp, total);
}

static inline void launch_pack_w(const float* W, int ldw, int colOff,
                                 int nSrcRows, int rowOff, __bf16* Wp,
                                 int K, int Kp, int nOutRows, hipStream_t s)
{
    int total = nOutRows * Kp;
    pack_w_kernel<<<(total + 255) / 256, 256, 0, s>>>(W, ldw, colOff, nSrcRows, rowOff,
                                                      Wp, K, Kp, total);
}

extern "C" void kernel_launch(void* const* d_in, const int* in_sizes, int n_in,
                              void* d_out, int out_size, void* d_ws, size_t ws_size,
                              hipStream_t stream)
{
    (void)in_sizes; (void)n_in; (void)out_size; (void)ws_size;
    const int Bb = 4, Nn = 4096, M = Bb * Nn;

    const float* xyz   = (const float*)d_in[0];
    const float* ec1_w = (const float*)d_in[1];   const float* ec1_bn[4] = {(const float*)d_in[2],(const float*)d_in[3],(const float*)d_in[4],(const float*)d_in[5]};
    const float* ec2_w = (const float*)d_in[6];   const float* ec2_bn[4] = {(const float*)d_in[7],(const float*)d_in[8],(const float*)d_in[9],(const float*)d_in[10]};
    const float* ec3_w = (const float*)d_in[11];  const float* ec3_bn[4] = {(const float*)d_in[12],(const float*)d_in[13],(const float*)d_in[14],(const float*)d_in[15]};
    const float* ec4_w = (const float*)d_in[16];  const float* ec4_bn[4] = {(const float*)d_in[17],(const float*)d_in[18],(const float*)d_in[19],(const float*)d_in[20]};
    const float* g_w   = (const float*)d_in[21];  const float* g_bn[4]   = {(const float*)d_in[22],(const float*)d_in[23],(const float*)d_in[24],(const float*)d_in[25]};
    const float* c1_w  = (const float*)d_in[26];  const float* c1_bn[4]  = {(const float*)d_in[27],(const float*)d_in[28],(const float*)d_in[29],(const float*)d_in[30]};
    const float* c2_w  = (const float*)d_in[31];  const float* c2_bn[4]  = {(const float*)d_in[32],(const float*)d_in[33],(const float*)d_in[34],(const float*)d_in[35]};
    const float* cls_w = (const float*)d_in[36];
    const float* cls_b = (const float*)d_in[37];

    float* ws = (float*)d_ws;
    const size_t OFF_XCAT = 0;                               // [M,512]  f32
    const size_t OFF_YZ   = OFF_XCAT + (size_t)M * 512;      // [M,1536] f32 (yz / g / feat)
    const size_t OFF_H1   = OFF_YZ   + (size_t)M * 1536;     // [M,256]  f32
    const size_t OFF_H2   = OFF_H1   + (size_t)M * 256;      // [M,256]  f32
    const size_t OFF_GMAX = OFF_H2   + (size_t)M * 256;      // [B,1024] f32
    const size_t OFF_IDX  = OFF_GMAX + (size_t)Bb * 1024;    // [M,20]   i32
    const size_t OFF_APK  = OFF_IDX  + (size_t)M * KNN_K;    // [M,1536] bf16 -> M*768 f32 slots
    const size_t OFF_WPK  = OFF_APK  + (size_t)M * 768;      // [1024,1536] bf16 max
    float* xcat = ws + OFF_XCAT;
    float* yzb  = ws + OFF_YZ;
    float* h1   = ws + OFF_H1;
    float* h2   = ws + OFF_H2;
    float* gmax = ws + OFF_GMAX;
    int*   idxb = (int*)(ws + OFF_IDX);
    __bf16* apk = (__bf16*)(ws + OFF_APK);
    __bf16* wpk = (__bf16*)(ws + OFF_WPK);

    const int knnBlocks = Bb * (Nn / 64);
    auto comb = [&](int Co, const float* const* bn, float* out) {
        int total = M * Co;
        edge_combine<<<(total + 255) / 256, 256, 0, stream>>>(
            yzb, idxb, bn[0], bn[1], bn[2], bn[3], out, 512, Nn, Co, total);
    };

    // ---- EdgeConv1: x=xyz (C=3, Kp=32) -> yz[M,128] -> x1(64) at xcat[:,0:64)
    knn_kernel<<<knnBlocks, 64, 0, stream>>>(xyz, 3, 3, Nn, idxb);
    launch_pack_a(xyz, 3, apk, 3, 32, M, stream);
    launch_pack_w(ec1_w, 6, 0, 64,  0, wpk, 3, 32, 64, stream);   // Wn rows
    launch_pack_w(ec1_w, 6, 3, 64, 64, wpk, 3, 32, 64, stream);   // Wc rows
    launch_gemm(apk, wpk, yzb, 128, M, 128, 128, 32,
                nullptr, nullptr, nullptr, nullptr, nullptr, 0, stream);
    comb(64, ec1_bn, xcat + 0);

    // ---- EdgeConv2: x=x1 (C=64) -> yz[M,128] -> x2(64) at xcat[:,64:128)
    knn_kernel<<<knnBlocks, 64, 0, stream>>>(xcat + 0, 512, 64, Nn, idxb);
    launch_pack_a(xcat, 512, apk, 64, 64, M, stream);
    launch_pack_w(ec2_w, 128,  0, 64,  0, wpk, 64, 64, 64, stream);
    launch_pack_w(ec2_w, 128, 64, 64, 64, wpk, 64, 64, 64, stream);
    launch_gemm(apk, wpk, yzb, 128, M, 128, 128, 64,
                nullptr, nullptr, nullptr, nullptr, nullptr, 0, stream);
    comb(64, ec2_bn, xcat + 64);

    // ---- EdgeConv3: x=x2 (C=64) -> yz[M,256] -> x3(128) at xcat[:,128:256)
    knn_kernel<<<knnBlocks, 64, 0, stream>>>(xcat + 64, 512, 64, Nn, idxb);
    launch_pack_a(xcat + 64, 512, apk, 64, 64, M, stream);
    launch_pack_w(ec3_w, 128,  0, 128,   0, wpk, 64, 64, 128, stream);
    launch_pack_w(ec3_w, 128, 64, 128, 128, wpk, 64, 64, 128, stream);
    launch_gemm(apk, wpk, yzb, 256, M, 256, 256, 64,
                nullptr, nullptr, nullptr, nullptr, nullptr, 0, stream);
    comb(128, ec3_bn, xcat + 128);

    // ---- EdgeConv4: x=x3 (C=128) -> yz[M,512] -> x4(256) at xcat[:,256:512)
    knn_kernel<<<knnBlocks, 64, 0, stream>>>(xcat + 128, 512, 128, Nn, idxb);
    launch_pack_a(xcat + 128, 512, apk, 128, 128, M, stream);
    launch_pack_w(ec4_w, 256,   0, 256,   0, wpk, 128, 128, 256, stream);
    launch_pack_w(ec4_w, 256, 128, 256, 256, wpk, 128, 128, 256, stream);
    launch_gemm(apk, wpk, yzb, 512, M, 512, 512, 128,
                nullptr, nullptr, nullptr, nullptr, nullptr, 0, stream);
    comb(256, ec4_bn, xcat + 256);

    // ---- Global feature: g = relu(bn(xcat . g_w^T)) [M,1024]; gmax = max over N
    launch_pack_a(xcat, 512, apk, 512, 512, M, stream);
    launch_pack_w(g_w, 512, 0, 1024, 0, wpk, 512, 512, 1024, stream);
    launch_gemm(apk, wpk, yzb, 1024, M, 1024, 1024, 512,
                g_bn[0], g_bn[1], g_bn[2], g_bn[3], nullptr, 1, stream);
    {
        int total = Bb * 1024;
        colmax_kernel<<<(total + 255) / 256, 256, 0, stream>>>(yzb, gmax, Nn, 1024, total);
    }
    {   // feat = [xcat | broadcast gmax] [M,1536]  (reuses yzb; g no longer needed)
        int total = M * 1536;
        build_feat_kernel<<<(total + 255) / 256, 256, 0, stream>>>(xcat, gmax, yzb, Nn, total);
    }

    // ---- Head: c1 -> c2 -> cls
    launch_pack_a(yzb, 1536, apk, 1536, 1536, M, stream);
    launch_pack_w(c1_w, 1536, 0, 256, 0, wpk, 1536, 1536, 256, stream);
    launch_gemm(apk, wpk, h1, 256, M, 256, 256, 1536,
                c1_bn[0], c1_bn[1], c1_bn[2], c1_bn[3], nullptr, 1, stream);

    launch_pack_a(h1, 256, apk, 256, 256, M, stream);
    launch_pack_w(c2_w, 256, 0, 256, 0, wpk, 256, 256, 256, stream);
    launch_gemm(apk, wpk, h2, 256, M, 256, 256, 256,
                c2_bn[0], c2_bn[1], c2_bn[2], c2_bn[3], nullptr, 1, stream);

    launch_pack_a(h2, 256, apk, 256, 256, M, stream);
    launch_pack_w(cls_w, 256, 0, 50, 0, wpk, 256, 256, 64, stream);
    launch_gemm(apk, wpk, (float*)d_out, 50, M, 50, 64, 256,
                nullptr, nullptr, nullptr, nullptr, cls_b, 0, stream);
}
